// Model_39676907887292
// MI455X (gfx1250) — matrix operand order repaired and checked by reading, not compiled, not run on hardware
//
#include <hip/hip_runtime.h>

// ---------------------------------------------------------------------------
// Fused BERT-style (quirky) attention block for MI455X (gfx1250, wave32).
//
//   qp = (q@Wq + bq) * (1/sqrt(256));  kp = k@Wk + bk;   (vp unused by ref!)
//   scores[b,h,s,j] = sum_m qp[b,s,h*128+m] * kp[b,m,h*128+j]   (no transpose)
//   probs = softmax(scores + (mask[b,j]==0 ? 1.0 : 0.0), axis=j)
//   out[b,s,:] = probs[b,s,(h,j)->flat] @ Wo + bo
//
// Design: memory-bound (384 MB min HBM traffic ~= 16.5us @ 23.3 TB/s vs only
// ~60 GFLOP), so fully fuse per batch: 1 workgroup (8 wave32) per b, all
// intermediates live in LDS as bf16 (QP 64KB, KP 64KB, 16KB staged weight
// chunk = 144KB -> 2 WGs/WGP of the 320KB budget). All matmuls use
// v_wmma_f32_16x16x32_bf16 with fp32 accumulation. Weight slabs are L2-
// resident; next slab is prefetched (global_prefetch_b8) while computing.
// ---------------------------------------------------------------------------

#define EMB   256
#define SEQ   128
#define NHEAD 2
#define DH    128

typedef __bf16 bf16_t;
typedef bf16_t v16bf __attribute__((ext_vector_type(16)));
typedef float  v8f   __attribute__((ext_vector_type(8)));

union BfFrag { v16bf v; unsigned short u[16]; unsigned int w[8]; };

// fp32 -> bf16 (RNE), scalar fallback
__device__ __forceinline__ unsigned short f2bf(float f) {
  union { float f; unsigned int u; } a; a.f = f;
  unsigned int u = a.u;
  unsigned int r = u + 0x7FFFu + ((u >> 16) & 1u);
  return (unsigned short)(r >> 16);
}

// two fp32 -> packed bf16x2 (lo in [15:0]); use HW packed convert if present
__device__ __forceinline__ unsigned int f2bf2(float lo, float hi) {
#if __has_builtin(__builtin_amdgcn_cvt_pk_bf16_f32)
  auto pk = __builtin_amdgcn_cvt_pk_bf16_f32(lo, hi);
  static_assert(sizeof(pk) == 4, "packed bf16x2 expected to be 32-bit");
  unsigned int bits;
  __builtin_memcpy(&bits, &pk, sizeof(bits));
  return bits;
#else
  return (unsigned int)f2bf(lo) | ((unsigned int)f2bf(hi) << 16);
#endif
}

__device__ __forceinline__ v8f wmma_bf16(v16bf a, v16bf b, v8f c) {
  // D = A(16x32 bf16) x B(32x16 bf16) + C(16x16 f32)
  return __builtin_amdgcn_wmma_f32_16x16x32_bf16(
      /*neg_a=*/false, a, /*neg_b=*/false, b,
      /*c_mod=*/(short)0, c, /*reuse_a=*/false, /*reuse_b=*/false);
}

// A fragment (16x32, MxK bf16) from an LDS bf16 matrix with row stride `ld`.
// ISA layout: lane%16 = M; lanes 0-15 hold K {0..7,16..23}, lanes 16-31 hold
// K {8..15,24..31}  (element e: K = 8*hi + (e&7) + 16*(e>>3)).
__device__ __forceinline__ v16bf load_a_lds(const unsigned short* base,
                                            int row0, int col0, int ld, int lane) {
  const int m  = lane & 15;
  const int hi = (lane >> 4) & 1;
  const unsigned short* p = base + (size_t)(row0 + m) * ld + col0 + 8 * hi;
  BfFrag f;
#pragma unroll
  for (int e = 0; e < 8; ++e) { f.u[e] = p[e]; f.u[e + 8] = p[e + 16]; }
  return f.v;
}

// Same A fragment sourced from a global fp32 matrix (packed convert).
__device__ __forceinline__ v16bf load_a_gf32(const float* base,
                                             int row0, int col0, int ld, int lane) {
  const int m  = lane & 15;
  const int hi = (lane >> 4) & 1;
  const float* p = base + (size_t)(row0 + m) * ld + col0 + 8 * hi;
  BfFrag f;
#pragma unroll
  for (int e = 0; e < 8; e += 2) {
    f.w[(e >> 1)]     = f2bf2(p[e],      p[e + 1]);
    f.w[(e >> 1) + 4] = f2bf2(p[e + 16], p[e + 17]);
  }
  return f.v;
}

// B fragment (32x16, KxN bf16) from LDS: lane%16 = N, element e: K = 8*hi + (e&7) + 16*(e>>3).
__device__ __forceinline__ v16bf load_b_lds(const unsigned short* base,
                                            int krow0, int col0, int ld, int lane) {
  const int hi = (lane >> 4) & 1;
  const unsigned short* p = base + (size_t)(krow0 + 8 * hi) * ld + col0 + (lane & 15);
  BfFrag f;
#pragma unroll
  for (int e = 0; e < 8; ++e) { f.u[e] = p[(size_t)e * ld]; f.u[e + 8] = p[(size_t)(e + 16) * ld]; }
  return f.v;
}

// Stage a fp32 [32][EMB] weight slab into LDS as bf16 (whole block helps),
// prefetching the next slab into L2/WGP caches.
__device__ __forceinline__ void stage_wslab(const float* W, int kc,
                                            unsigned short* WC, int tid,
                                            bool prefetch_next) {
  const int row = tid >> 3;           // 0..31
  const int c0  = (tid & 7) * 32;     // 0..224
  const float* wp = W + (size_t)(32 * kc + row) * EMB + c0;
  unsigned short* op = WC + row * EMB + c0;
#pragma unroll
  for (int i = 0; i < 32; i += 2)
    *reinterpret_cast<unsigned int*>(op + i) = f2bf2(wp[i], wp[i + 1]);
  if (prefetch_next)
    __builtin_prefetch(wp + 32 * EMB, 0, 0);   // -> global_prefetch_b8
}

__global__ __launch_bounds__(256) void fused_attn_kernel(
    const float* __restrict__ q, const float* __restrict__ k,
    const int* __restrict__ amask,
    const float* __restrict__ Wq, const float* __restrict__ bq,
    const float* __restrict__ Wk, const float* __restrict__ bk,
    const float* __restrict__ Wo, const float* __restrict__ bo,
    float* __restrict__ out) {
  extern __shared__ unsigned short smem[];          // bf16 stored as ushort
  unsigned short* QP = smem;                        // [128][256] qp -> probs
  unsigned short* KP = smem + SEQ * EMB;            // [128][256] kp
  unsigned short* WC = smem + 2 * SEQ * EMB;        // [32][256] weight slab

  const int b    = blockIdx.x;
  const int tid  = threadIdx.x;
  const int lane = tid & 31;
  const int wave = tid >> 5;        // 0..7 : 16-row output band per wave
  const int l16  = lane & 15;
  const int hi   = (lane >> 4) & 1;

  const float* qb = q + (size_t)b * SEQ * EMB;
  const float* kb = k + (size_t)b * SEQ * EMB;
  const v8f zacc = {};

  // ================= Stage 1: QP = (q@Wq+bq)/16, KP = k@Wk+bk ==============
  for (int proj = 0; proj < 2; ++proj) {
    const float* src    = (proj == 0) ? qb : kb;
    const float* W      = (proj == 0) ? Wq : Wk;
    const float* bias   = (proj == 0) ? bq : bk;
    unsigned short* dst = (proj == 0) ? QP : KP;
    const float scale   = (proj == 0) ? 0.0625f : 1.0f;   // 1/sqrt(256)

    v8f acc[16];
#pragma unroll
    for (int ni = 0; ni < 16; ++ni) acc[ni] = zacc;

    for (int kc = 0; kc < 8; ++kc) {          // K = 256 in slabs of 32
      stage_wslab(W, kc, WC, tid, kc < 7);
      __syncthreads();
      v16bf afrag = load_a_gf32(src, 16 * wave, 32 * kc, EMB, lane);
      if (kc < 7)  // prefetch next K-slab of the activation row
        __builtin_prefetch(src + (size_t)(16 * wave + l16) * EMB + 32 * (kc + 1), 0, 0);
#pragma unroll
      for (int ni = 0; ni < 16; ++ni) {
        v16bf bfrag = load_b_lds(WC, 0, 16 * ni, EMB, lane);
        acc[ni] = wmma_bf16(afrag, bfrag, acc[ni]);
      }
      __syncthreads();                         // WAR on WC before next slab
    }
    // write this wave's 16-row band (C/D layout: M = r + 8*hi, N = l16)
#pragma unroll
    for (int ni = 0; ni < 16; ++ni) {
      const int n  = 16 * ni + l16;
      const float bb = bias[n];
#pragma unroll
      for (int r = 0; r < 8; ++r) {
        const int m = 16 * wave + r + 8 * hi;
        dst[(size_t)m * EMB + n] = f2bf((acc[ni][r] + bb) * scale);
      }
    }
    __syncthreads();                           // KP visible to all waves
  }

  // ============ Stage 2: scores = QPh @ KPh, +mask, softmax -> probs =======
  // mask add depends only on column j = 16*nj + l16 (reference quirk: +1.0
  // where mask==0, added along the *head-dim* axis which happens to be S)
  float madd[8];
#pragma unroll
  for (int nj = 0; nj < 8; ++nj)
    madd[nj] = (amask[(size_t)b * SEQ + 16 * nj + l16] == 0) ? 1.0f : 0.0f;

  for (int h = 0; h < NHEAD; ++h) {
    const int c0 = h * DH;
    v8f acc[8];
#pragma unroll
    for (int nj = 0; nj < 8; ++nj) acc[nj] = zacc;
#pragma unroll
    for (int kc = 0; kc < 4; ++kc) {           // contraction dim = 128
      v16bf afrag = load_a_lds(QP, 16 * wave, c0 + 32 * kc, EMB, lane);
#pragma unroll
      for (int nj = 0; nj < 8; ++nj) {
        v16bf bfrag = load_b_lds(KP, 32 * kc, c0 + 16 * nj, EMB, lane);
        acc[nj] = wmma_bf16(afrag, bfrag, acc[nj]);
      }
    }
    // Row softmax. Row M = 16*wave + r + 8*hi lives entirely in one 16-lane
    // half of the wave -> reduce with shfl_xor over width 16.
#pragma unroll
    for (int r = 0; r < 8; ++r) {
      float x[8];
      float mx = -3.402823466e38f;
#pragma unroll
      for (int nj = 0; nj < 8; ++nj) { x[nj] = acc[nj][r] + madd[nj]; mx = fmaxf(mx, x[nj]); }
#pragma unroll
      for (int off = 8; off >= 1; off >>= 1) mx = fmaxf(mx, __shfl_xor(mx, off, 16));
      float sum = 0.0f;
#pragma unroll
      for (int nj = 0; nj < 8; ++nj) { x[nj] = __expf(x[nj] - mx); sum += x[nj]; }
#pragma unroll
      for (int off = 8; off >= 1; off >>= 1) sum += __shfl_xor(sum, off, 16);
      const float inv = 1.0f / sum;
      const int m = 16 * wave + r + 8 * hi;
      // probs already in [s, h*128+j] layout -> overwrite QP in place
      // (safe: this wave read its own QP band for this head above; head 1
      //  columns are untouched until its own pass)
#pragma unroll
      for (int nj = 0; nj < 8; ++nj)
        QP[(size_t)m * EMB + c0 + 16 * nj + l16] = f2bf(x[nj] * inv);
    }
  }
  __syncthreads();

  // ================= Stage 3: out = probs @ Wo + bo (fp32 out) =============
  {
    v8f acc[16];
#pragma unroll
    for (int ni = 0; ni < 16; ++ni) acc[ni] = zacc;
    for (int kc = 0; kc < 8; ++kc) {
      stage_wslab(Wo, kc, WC, tid, kc < 7);
      __syncthreads();
      v16bf afrag = load_a_lds(QP /*probs*/, 16 * wave, 32 * kc, EMB, lane);
#pragma unroll
      for (int ni = 0; ni < 16; ++ni) {
        v16bf bfrag = load_b_lds(WC, 0, 16 * ni, EMB, lane);
        acc[ni] = wmma_bf16(afrag, bfrag, acc[ni]);
      }
      __syncthreads();
    }
    float* ob = out + (size_t)b * SEQ * EMB;
#pragma unroll
    for (int ni = 0; ni < 16; ++ni) {
      const int n  = 16 * ni + l16;
      const float bb = bo[n];
#pragma unroll
      for (int r = 0; r < 8; ++r) {
        const int m = 16 * wave + r + 8 * hi;
        ob[(size_t)m * EMB + n] = acc[ni][r] + bb;
      }
    }
  }
}

extern "C" void kernel_launch(void* const* d_in, const int* in_sizes, int n_in,
                              void* d_out, int out_size, void* d_ws, size_t ws_size,
                              hipStream_t stream) {
  (void)in_sizes; (void)n_in; (void)out_size; (void)d_ws; (void)ws_size;
  const float* q   = (const float*)d_in[0];
  const float* k   = (const float*)d_in[1];
  // d_in[2] = v : projected but unused downstream in the reference -> skipped
  const int*  amsk = (const int*)d_in[3];
  const float* Wq  = (const float*)d_in[4];
  const float* bq  = (const float*)d_in[5];
  const float* Wk  = (const float*)d_in[6];
  const float* bk  = (const float*)d_in[7];
  // d_in[8] = Wv, d_in[9] = bv : unused
  const float* Wo  = (const float*)d_in[10];
  const float* bo  = (const float*)d_in[11];
  float* out = (float*)d_out;

  const size_t lds_bytes =
      (size_t)(2 * SEQ * EMB + 32 * EMB) * sizeof(unsigned short);  // 144 KB
  hipFuncSetAttribute(reinterpret_cast<const void*>(fused_attn_kernel),
                      hipFuncAttributeMaxDynamicSharedMemorySize,
                      (int)lds_bytes);
  fused_attn_kernel<<<1024, 256, lds_bytes, stream>>>(
      q, k, amsk, Wq, bq, Wk, bk, Wo, bo, out);
}
